// SupervisorGating_51410758533302
// MI455X (gfx1250) — matrix operand order repaired and checked by reading, not compiled, not run on hardware
//
#include <hip/hip_runtime.h>
#include <hip/hip_bf16.h>

// ---------------------------------------------------------------------------
// SupervisorGating fused pipeline for MI455X (gfx1250, wave32, WMMA bf16)
//
//   scores = x @ Mfold + c0          (Mfold folds wq and k-projection, /8)
//   attn   = softmax_per_head(scores)
//   h1     = gelu(attn @ G1 + g1b)   (G1 folds v-proj, out_proj and gate_w1)
//   logits = h1 @ W2^T + b2
//   out    = top2(softmax(logits)), renormalized
//
// Wave tiling: each wave owns 2 m-tiles x 4 n-tiles so every B fragment
// (streamed from the 192MB L2) feeds two v_wmma ops.
// ---------------------------------------------------------------------------

typedef __attribute__((ext_vector_type(16))) __bf16 v16bf;
typedef __attribute__((ext_vector_type(8)))  __bf16 v8bf;
typedef __attribute__((ext_vector_type(8)))  float  v8f;

#define D_MODEL 1024
#define N_HEAD  16
#define HDIM    64
#define N_EXP   64
#define N_G1    256      // 4*E
#define M_TILE  64       // tokens per block
#define XSTR    1032     // bf16 LDS row stride for 1024-wide tiles (16B-aligned, padded)
#define H1STR   264      // bf16 LDS row stride for 256-wide tile
#define LGSTR   65       // f32 LDS row stride for logits

__device__ inline v8f zero8() {
  v8f z;
#pragma unroll
  for (int i = 0; i < 8; ++i) z[i] = 0.0f;
  return z;
}

// A-fragment (16x32 bf16): lane<16 -> K 0..15, lane>=16 -> K 16..31
__device__ inline v16bf load_a_frag(const __bf16* buf, int stride, int mtile, int k, int lane) {
  const int m    = lane & 15;
  const int koff = (lane < 16) ? 0 : 16;
  const __bf16* p = buf + (mtile * 16 + m) * stride + k + koff;
  v8bf lo = *reinterpret_cast<const v8bf*>(p);        // K = k+koff .. +7
  v8bf hi = *reinterpret_cast<const v8bf*>(p + 8);    // K = k+koff+8 .. +15
  v16bf r;
#pragma unroll
  for (int i = 0; i < 8; ++i) { r[i] = lo[i]; r[i + 8] = hi[i]; }
  return r;
}

// B-fragment (32x16 bf16): B stored row-major as (N, K); lane<16 -> K 0..15, lane>=16 -> K 16..31
__device__ inline v16bf load_b_frag(const __bf16* B, int Kdim, int n, int k, int lane) {
  const int nl = lane & 15;
  const int kb = (lane < 16) ? 0 : 16;
  const __bf16* p = B + (size_t)(n + nl) * Kdim + k + kb;
  return *reinterpret_cast<const v16bf*>(p);          // 32B contiguous per lane
}

__device__ inline v8f wmma_bf16(v16bf a, v16bf b, v8f c) {
  return __builtin_amdgcn_wmma_f32_16x16x32_bf16(false, a, false, b, (short)0, c, false, false);
}

// ------------------------------- precompute --------------------------------

// k/v projections: out[e,c] = emb[e,:] . W[c,:] + b[c]   (W = wk or wv rows of in_proj)
__global__ void k_proj_kv(const float* __restrict__ emb, const float* __restrict__ w_in,
                          const float* __restrict__ b_in, float* __restrict__ kout,
                          float* __restrict__ vout) {
  int idx = blockIdx.x * blockDim.x + threadIdx.x;   // < 2*E*D
  int kv  = idx >> 16;
  int rem = idx & 65535;
  int e = rem >> 10, c = rem & 1023;
  int row = (1 + kv) * D_MODEL + c;                  // wk rows D.., wv rows 2D..
  const float4* a = (const float4*)(emb + (size_t)e * D_MODEL);
  const float4* w = (const float4*)(w_in + (size_t)row * D_MODEL);
  float s = 0.f;
  for (int i = 0; i < D_MODEL / 4; ++i) {
    float4 av = a[i], wv4 = w[i];
    s += av.x * wv4.x + av.y * wv4.y + av.z * wv4.z + av.w * wv4.w;
  }
  s += b_in[row];
  (kv == 0 ? kout : vout)[(size_t)e * D_MODEL + c] = s;
}

// Mfold^T[n=(h,e), i] = (1/8) sum_j wq[h*64+j, i] * k[e, h*64+j]
__global__ void k_fold_M(const float* __restrict__ wq, const float* __restrict__ kproj,
                         __bf16* __restrict__ Mt) {
  int idx = blockIdx.x * blockDim.x + threadIdx.x;   // < D*D
  int n = idx >> 10, i = idx & 1023;
  int h = n >> 6, e = n & 63;
  const float* kr = kproj + (size_t)e * D_MODEL + h * HDIM;
  float s = 0.f;
#pragma unroll 4
  for (int j = 0; j < HDIM; ++j) s += wq[(size_t)(h * HDIM + j) * D_MODEL + i] * kr[j];
  Mt[(size_t)n * D_MODEL + i] = (__bf16)(s * 0.125f);
}

// c0[(h,e)] = (1/8) sum_j bq[h*64+j] * k[e, h*64+j]
__global__ void k_fold_c0(const float* __restrict__ b_in, const float* __restrict__ kproj,
                          float* __restrict__ c0) {
  int n = blockIdx.x * blockDim.x + threadIdx.x;     // < 1024
  int h = n >> 6, e = n & 63;
  float s = 0.f;
  for (int j = 0; j < HDIM; ++j) s += b_in[h * HDIM + j] * kproj[(size_t)e * D_MODEL + h * HDIM + j];
  c0[n] = s * 0.125f;
}

// Pfold[(h,e), m] = sum_j v[e, h*64+j] * wout[m, h*64+j]
__global__ void k_fold_P(const float* __restrict__ vproj, const float* __restrict__ wout,
                         float* __restrict__ Pf) {
  int idx = blockIdx.x * blockDim.x + threadIdx.x;   // < D*D
  int he = idx >> 10, m = idx & 1023;
  int h = he >> 6, e = he & 63;
  const float* vr = vproj + (size_t)e * D_MODEL + h * HDIM;
  const float* wr = wout + (size_t)m * D_MODEL + h * HDIM;
  float s = 0.f;
#pragma unroll 4
  for (int j = 0; j < HDIM; ++j) s += vr[j] * wr[j];
  Pf[(size_t)he * D_MODEL + m] = s;
}

// G1^T[n, he] = sum_m Pfold[he, m] * W1[n, m]
__global__ void k_fold_G1(const float* __restrict__ Pf, const float* __restrict__ w1,
                          __bf16* __restrict__ G1t) {
  int idx = blockIdx.x * blockDim.x + threadIdx.x;   // < 256*1024
  int n = idx >> 10, he = idx & 1023;
  const float4* p = (const float4*)(Pf + (size_t)he * D_MODEL);
  const float4* w = (const float4*)(w1 + (size_t)n * D_MODEL);
  float s = 0.f;
  for (int i = 0; i < D_MODEL / 4; ++i) {
    float4 a = p[i], b = w[i];
    s += a.x * b.x + a.y * b.y + a.z * b.z + a.w * b.w;
  }
  G1t[(size_t)n * D_MODEL + he] = (__bf16)s;
}

// g1b[n] = b1[n] + sum_m bout[m] * W1[n, m]
__global__ void k_fold_g1b(const float* __restrict__ b1, const float* __restrict__ bout,
                           const float* __restrict__ w1, float* __restrict__ g1b) {
  int n = threadIdx.x;                               // 256 threads
  const float4* bo = (const float4*)bout;
  const float4* w = (const float4*)(w1 + (size_t)n * D_MODEL);
  float s = 0.f;
  for (int i = 0; i < D_MODEL / 4; ++i) {
    float4 a = bo[i], b = w[i];
    s += a.x * b.x + a.y * b.y + a.z * b.z + a.w * b.w;
  }
  g1b[n] = s + b1[n];
}

__global__ void k_cvt_w2(const float* __restrict__ w2, __bf16* __restrict__ w2b) {
  int idx = blockIdx.x * blockDim.x + threadIdx.x;   // < 64*256
  w2b[idx] = (__bf16)w2[idx];
}

// ------------------------------- main kernel -------------------------------

__global__ __launch_bounds__(256) void k_gating_main(
    const float* __restrict__ x, const __bf16* __restrict__ Mt,
    const float* __restrict__ c0, const __bf16* __restrict__ G1t,
    const float* __restrict__ g1b, const __bf16* __restrict__ W2b,
    const float* __restrict__ b2, int* __restrict__ out_idx,
    float* __restrict__ out_w) {
  extern __shared__ __align__(32) char smem[];
  __bf16* x_lds    = (__bf16*)smem;                        // 64*1032*2 = 132096 B
  __bf16* attn_lds = (__bf16*)(smem + M_TILE * XSTR * 2);  // 132096 B
  __bf16* h1_lds   = (__bf16*)smem;                        // aliases x (dead after phase 1)
  float*  lg_lds   = (float*)(smem + 36864);               // aliases x tail

  const int tid   = threadIdx.x;
  const int lane  = tid & 31;
  const int wave  = tid >> 5;
  const int mpair = wave >> 2;       // 0..1 -> m-tiles mpair*2, mpair*2+1
  const int nq    = wave & 3;        // 0..3 -> 4 n-tiles (one head per 256-col chunk)
  const int nl    = lane & 15;
  const int lhi   = (lane < 16) ? 0 : 8;

  // ---- phase 0: stage x tile -> LDS as bf16 -------------------------------
  {
    const float4* x4 = (const float4*)(x + (size_t)blockIdx.x * M_TILE * D_MODEL);
    for (int i = tid; i < M_TILE * (D_MODEL / 4); i += 256) {
      int row = i >> 8, c4 = i & 255;
      float4 v = x4[row * (D_MODEL / 4) + c4];
      __bf16* dst = x_lds + row * XSTR + c4 * 4;
      dst[0] = (__bf16)v.x; dst[1] = (__bf16)v.y; dst[2] = (__bf16)v.z; dst[3] = (__bf16)v.w;
    }
  }
  __syncthreads();

  // ---- phase 1: scores = x @ Mfold + c0, fused per-head softmax -> attn ---
  for (int ch = 0; ch < 4; ++ch) {
    v8f acc[2][4];
#pragma unroll
    for (int mi = 0; mi < 2; ++mi)
#pragma unroll
      for (int j = 0; j < 4; ++j) acc[mi][j] = zero8();
    const int hbase = ch * 256 + nq * 64;            // this wave's head (64 cols)
    for (int k = 0; k < D_MODEL; k += 32) {
      v16bf a0 = load_a_frag(x_lds, XSTR, mpair * 2 + 0, k, lane);
      v16bf a1 = load_a_frag(x_lds, XSTR, mpair * 2 + 1, k, lane);
      __builtin_prefetch(Mt + (size_t)(hbase + nl) * D_MODEL + k + 32, 0, 1);
#pragma unroll
      for (int j = 0; j < 4; ++j) {
        v16bf b = load_b_frag(Mt, D_MODEL, hbase + j * 16, k, lane);
        acc[0][j] = wmma_bf16(a0, b, acc[0][j]);
        acc[1][j] = wmma_bf16(a1, b, acc[1][j]);
      }
    }
#pragma unroll
    for (int j = 0; j < 4; ++j) {
      float c0v = c0[hbase + j * 16 + nl];
#pragma unroll
      for (int mi = 0; mi < 2; ++mi)
#pragma unroll
        for (int r = 0; r < 8; ++r) acc[mi][j][r] += c0v;
    }
    // softmax over the wave's head (64 cols = 4 n-tiles x 16 lanes)
#pragma unroll
    for (int mi = 0; mi < 2; ++mi) {
      const int mb = (mpair * 2 + mi) * 16 + lhi;
#pragma unroll
      for (int r = 0; r < 8; ++r) {
        float mx = acc[mi][0][r];
#pragma unroll
        for (int j = 1; j < 4; ++j) mx = fmaxf(mx, acc[mi][j][r]);
        for (int off = 1; off < 16; off <<= 1) mx = fmaxf(mx, __shfl_xor(mx, off, 32));
        float e0 = __expf(acc[mi][0][r] - mx);
        float e1 = __expf(acc[mi][1][r] - mx);
        float e2 = __expf(acc[mi][2][r] - mx);
        float e3 = __expf(acc[mi][3][r] - mx);
        float s = e0 + e1 + e2 + e3;
        for (int off = 1; off < 16; off <<= 1) s += __shfl_xor(s, off, 32);
        float inv = 1.0f / s;
        __bf16* arow = attn_lds + (mb + r) * XSTR + hbase + nl;
        arow[0]  = (__bf16)(e0 * inv);
        arow[16] = (__bf16)(e1 * inv);
        arow[32] = (__bf16)(e2 * inv);
        arow[48] = (__bf16)(e3 * inv);
      }
    }
  }
  __syncthreads();   // attn complete; x region now dead -> reused for h1/logits

  // ---- phase 3: h1 = gelu(attn @ G1 + g1b) --------------------------------
  {
    v8f acc[2][4];
#pragma unroll
    for (int mi = 0; mi < 2; ++mi)
#pragma unroll
      for (int j = 0; j < 4; ++j) acc[mi][j] = zero8();
    for (int k = 0; k < D_MODEL; k += 32) {
      v16bf a0 = load_a_frag(attn_lds, XSTR, mpair * 2 + 0, k, lane);
      v16bf a1 = load_a_frag(attn_lds, XSTR, mpair * 2 + 1, k, lane);
      __builtin_prefetch(G1t + (size_t)(nq * 64 + nl) * D_MODEL + k + 32, 0, 1);
#pragma unroll
      for (int j = 0; j < 4; ++j) {
        v16bf b = load_b_frag(G1t, D_MODEL, (nq * 4 + j) * 16, k, lane);
        acc[0][j] = wmma_bf16(a0, b, acc[0][j]);
        acc[1][j] = wmma_bf16(a1, b, acc[1][j]);
      }
    }
#pragma unroll
    for (int j = 0; j < 4; ++j) {
      int col = (nq * 4 + j) * 16 + nl;
      float bias = g1b[col];
#pragma unroll
      for (int mi = 0; mi < 2; ++mi) {
        const int mb = (mpair * 2 + mi) * 16 + lhi;
#pragma unroll
        for (int r = 0; r < 8; ++r) {
          float v = acc[mi][j][r] + bias;
          float g = 0.5f * v * (1.0f + erff(v * 0.70710678118654752f));  // exact GELU
          h1_lds[(mb + r) * H1STR + col] = (__bf16)g;
        }
      }
    }
  }
  __syncthreads();

  // ---- phase 4: logits = h1 @ W2^T + b2 -----------------------------------
  {
    v8f acc[2];
    acc[0] = zero8(); acc[1] = zero8();
    for (int k = 0; k < N_G1; k += 32) {
      v16bf a0 = load_a_frag(h1_lds, H1STR, mpair * 2 + 0, k, lane);
      v16bf a1 = load_a_frag(h1_lds, H1STR, mpair * 2 + 1, k, lane);
      v16bf b  = load_b_frag(W2b, N_G1, nq * 16, k, lane);
      acc[0] = wmma_bf16(a0, b, acc[0]);
      acc[1] = wmma_bf16(a1, b, acc[1]);
    }
    int col = nq * 16 + nl;
    float bias = b2[col];
#pragma unroll
    for (int mi = 0; mi < 2; ++mi) {
      const int mb = (mpair * 2 + mi) * 16 + lhi;
#pragma unroll
      for (int r = 0; r < 8; ++r)
        lg_lds[(mb + r) * LGSTR + col] = acc[mi][r] + bias;
    }
  }
  __syncthreads();

  // ---- phase 5: softmax over 64 experts, top-2, renormalize ---------------
  if (tid < M_TILE) {
    const float* L = lg_lds + tid * LGSTR;
    float mx = L[0];
    float l1 = -1e30f, l2 = -1e30f;
    int i1 = 0, i2 = 0;
    for (int i = 0; i < N_EXP; ++i) {
      float v = L[i];
      mx = fmaxf(mx, v);
      if (v > l1)      { l2 = l1; i2 = i1; l1 = v; i1 = i; }
      else if (v > l2) { l2 = v; i2 = i; }
    }
    float s = 0.f;
    for (int i = 0; i < N_EXP; ++i) s += __expf(L[i] - mx);
    float inv = 1.0f / s;
    float p1 = __expf(l1 - mx) * inv;
    float p2 = __expf(l2 - mx) * inv;
    float wsum = 1.0f / (p1 + p2 + 1e-8f);
    size_t t = (size_t)blockIdx.x * M_TILE + tid;
    out_idx[t * 2 + 0] = i1;
    out_idx[t * 2 + 1] = i2;
    out_w[t * 2 + 0] = p1 * wsum;
    out_w[t * 2 + 1] = p2 * wsum;
  }
}

// ------------------------------- host glue ---------------------------------

extern "C" void kernel_launch(void* const* d_in, const int* in_sizes, int n_in,
                              void* d_out, int out_size, void* d_ws, size_t ws_size,
                              hipStream_t stream) {
  const float* x    = (const float*)d_in[0];
  const float* emb  = (const float*)d_in[1];
  const float* winp = (const float*)d_in[2];
  const float* binp = (const float*)d_in[3];
  const float* wout = (const float*)d_in[4];
  const float* bout = (const float*)d_in[5];
  const float* gw1  = (const float*)d_in[6];
  const float* gb1  = (const float*)d_in[7];
  const float* gw2  = (const float*)d_in[8];
  const float* gb2  = (const float*)d_in[9];

  const int n_tok = in_sizes[0] / D_MODEL;   // 16384

  char* ws = (char*)d_ws;
  size_t off = 0;
  auto alloc = [&](size_t bytes) { char* p = ws + off; off = (off + bytes + 255) & ~(size_t)255; return p; };
  float*  kproj = (float*)alloc((size_t)N_EXP * D_MODEL * 4);
  float*  vproj = (float*)alloc((size_t)N_EXP * D_MODEL * 4);
  __bf16* Mt    = (__bf16*)alloc((size_t)D_MODEL * D_MODEL * 2);
  float*  Pf    = (float*)alloc((size_t)D_MODEL * D_MODEL * 4);
  __bf16* G1t   = (__bf16*)alloc((size_t)N_G1 * D_MODEL * 2);
  __bf16* W2b   = (__bf16*)alloc((size_t)N_EXP * N_G1 * 2);
  float*  c0    = (float*)alloc((size_t)D_MODEL * 4);
  float*  g1b   = (float*)alloc((size_t)N_G1 * 4);

  k_proj_kv<<<(2 * N_EXP * D_MODEL) / 256, 256, 0, stream>>>(emb, winp, binp, kproj, vproj);
  k_fold_M<<<(D_MODEL * D_MODEL) / 256, 256, 0, stream>>>(winp, kproj, Mt);
  k_fold_c0<<<4, 256, 0, stream>>>(binp, kproj, c0);
  k_fold_P<<<(D_MODEL * D_MODEL) / 256, 256, 0, stream>>>(vproj, wout, Pf);
  k_fold_G1<<<(N_G1 * D_MODEL) / 256, 256, 0, stream>>>(Pf, gw1, G1t);
  k_fold_g1b<<<1, 256, 0, stream>>>(gb1, bout, gw1, g1b);
  k_cvt_w2<<<(N_EXP * N_G1) / 256, 256, 0, stream>>>(gw2, W2b);

  int* out_idx = (int*)d_out;
  float* out_w = (float*)d_out + (size_t)n_tok * 2;
  const size_t smem = (size_t)(2 * M_TILE * XSTR * 2);   // 264192 B (<= 320KB WGP LDS)
  k_gating_main<<<n_tok / M_TILE, 256, smem, stream>>>(x, Mt, c0, G1t, g1b, W2b, gb2,
                                                       out_idx, out_w);
}
